// GraphSAGEModel_43542378447394
// MI455X (gfx1250) — matrix operand order repaired and checked by reading, not compiled, not run on hardware
//
#include <hip/hip_runtime.h>

typedef __attribute__((ext_vector_type(16))) _Float16 v16h;
typedef __attribute__((ext_vector_type(8)))  _Float16 v8h;
typedef __attribute__((ext_vector_type(4)))  _Float16 v4h;
typedef __attribute__((ext_vector_type(8)))  float    v8f;

#define NNODE 50000
#define NEDGE 1600000
#define INC   128

// ---------------------------------------------------------------------------
// Elementwise helpers
// ---------------------------------------------------------------------------
__global__ void zero_kernel(float* __restrict__ p, long long n) {
  long long i = (long long)blockIdx.x * blockDim.x + threadIdx.x;
  if (i < n) p[i] = 0.0f;
}

__global__ void cvt_f16_kernel(const float* __restrict__ in,
                               _Float16* __restrict__ out, long long n) {
  long long i = (long long)blockIdx.x * blockDim.x + threadIdx.x;
  if (i < n) out[i] = (_Float16)in[i];
}

// aggh = f16(agg * deg_inv[row]) : fuses mean normalization into conversion
__global__ void cvt_agg_kernel(const float* __restrict__ agg,
                               const float* __restrict__ dinv,
                               _Float16* __restrict__ out, long long n) {
  long long i = (long long)blockIdx.x * blockDim.x + threadIdx.x;
  if (i < n) out[i] = (_Float16)(agg[i] * dinv[i >> 7]);   // INC == 128
}

__global__ void degree_kernel(const long long* __restrict__ dst,
                              float* __restrict__ deg, int e) {
  int i = blockIdx.x * blockDim.x + threadIdx.x;
  if (i < e) atomicAdd(&deg[(int)dst[i]], 1.0f);
}

__global__ void deginv_kernel(float* __restrict__ deg, int n) {
  int i = blockIdx.x * blockDim.x + threadIdx.x;
  if (i < n) {
    float d = deg[i];
    deg[i] = (d > 0.0f) ? (1.0f / d) : 0.0f;   // matches reference where()
  }
}

// ---------------------------------------------------------------------------
// Edge scatter-add from f16 features: one 32-lane group per edge, each lane
// gathers 4 halfs (one b64) and does 4 f32 atomics (resolve in 192MB L2).
// ---------------------------------------------------------------------------
__global__ void scatter_h16_kernel(const _Float16* __restrict__ feat,
                                   const long long* __restrict__ src,
                                   const long long* __restrict__ dst,
                                   float* __restrict__ agg, int e) {
  int t    = blockIdx.x * blockDim.x + threadIdx.x;
  int edge = t >> 5;
  int lane = t & 31;
  if (edge < e) {
    long long s = src[edge];
    long long d = dst[edge];
    const v4h v = *(const v4h*)(feat + s * INC + lane * 4);
    float* a = agg + d * INC + lane * 4;
    atomicAdd(a + 0, (float)v[0]);
    atomicAdd(a + 1, (float)v[1]);
    atomicAdd(a + 2, (float)v[2]);
    atomicAdd(a + 3, (float)v[3]);
  }
}

// ---------------------------------------------------------------------------
// Fused SAGE linear from pre-converted f16 operands:
//   out = aggh @ Wl^T + bias + xh @ Wr^T   [+ReLU]
// One wave -> 16x16 f32 tile, K=128 in 4 steps, 2 WMMAs/step into one C.
// Inner loop is pure b128 loads + v_wmma_f32_16x16x32_f16.
// ---------------------------------------------------------------------------
union V16 { v16h v; v8h h[2]; };

__global__ __launch_bounds__(256)
void sage_wmma_kernel(const _Float16* __restrict__ aggh,
                      const _Float16* __restrict__ xh,
                      const _Float16* __restrict__ Wlh,  // [outc, INC] f16
                      const float*    __restrict__ bl,   // [outc]
                      const _Float16* __restrict__ Wrh,  // [outc, INC] f16
                      float*    __restrict__ outf,       // f32 out or nullptr
                      _Float16* __restrict__ outh,       // f16 out or nullptr
                      int outc, int do_relu) {
  const int lane  = threadIdx.x & 31;
  const int wave  = threadIdx.x >> 5;
  const int hi    = lane >> 4;        // 0: lanes 0-15, 1: lanes 16-31
  const int m15   = lane & 15;
  const int tileM = blockIdx.x * 16;  // N = 50000 = 3125*16, exact
  const int tileN = wave * 16;

  const int arow = tileM + m15;       // A row this lane loads
  const int bcol = tileN + m15;       // B column (= W row) this lane loads

  const _Float16* __restrict__ agg_row = aggh + (long long)arow * INC;
  const _Float16* __restrict__ x_row   = xh   + (long long)arow * INC;
  const _Float16* __restrict__ wl_row  = Wlh  + (long long)bcol * INC;
  const _Float16* __restrict__ wr_row  = Wrh  + (long long)bcol * INC;

  v8f c = {};
  #pragma unroll
  for (int kb = 0; kb < INC; kb += 32) {
    // A 16x32 f16: lanes 0-15 K-offsets {0..15}; lanes 16-31 {16..31} (contiguous)
    const int ka = kb + (hi ? 16 : 0);
    V16 a_agg, a_x;
    a_agg.h[0] = *(const v8h*)(agg_row + ka);
    a_agg.h[1] = *(const v8h*)(agg_row + ka + 8);
    a_x.h[0]   = *(const v8h*)(x_row + ka);
    a_x.h[1]   = *(const v8h*)(x_row + ka + 8);
    // B 32x16 f16: lanes 0-15 K=kb+0..15; lanes 16-31 K=kb+16..31 (contiguous)
    const int kbb = kb + (hi ? 16 : 0);
    const v16h b_l = *(const v16h*)(wl_row + kbb);
    const v16h b_r = *(const v16h*)(wr_row + kbb);
    c = __builtin_amdgcn_wmma_f32_16x16x32_f16(false, a_agg.v, false, b_l,
                                               (short)0, c, false, false);
    c = __builtin_amdgcn_wmma_f32_16x16x32_f16(false, a_x.v,   false, b_r,
                                               (short)0, c, false, false);
  }

  // C/D f32 layout: VGPR r -> M=r (lanes 0-15) / M=8+r (lanes 16-31), N=lane&15
  const float bias  = bl[bcol];
  const int   mbase = tileM + hi * 8;
  if (outh) {
    #pragma unroll
    for (int r = 0; r < 8; ++r) {
      float v = c[r] + bias;
      if (do_relu) v = fmaxf(v, 0.0f);
      outh[(long long)(mbase + r) * outc + bcol] = (_Float16)v;
    }
  } else {
    #pragma unroll
    for (int r = 0; r < 8; ++r) {
      float v = c[r] + bias;
      if (do_relu) v = fmaxf(v, 0.0f);
      outf[(long long)(mbase + r) * outc + bcol] = v;
    }
  }
}

// ---------------------------------------------------------------------------
// Launcher
// ---------------------------------------------------------------------------
extern "C" void kernel_launch(void* const* d_in, const int* in_sizes, int n_in,
                              void* d_out, int out_size, void* d_ws, size_t ws_size,
                              hipStream_t stream) {
  const float*     x    = (const float*)d_in[0];
  const long long* ei   = (const long long*)d_in[1];   // int64 [2, E]
  const long long* src  = ei;
  const long long* dstp = ei + NEDGE;
  const float* Wl1 = (const float*)d_in[2];
  const float* bl1 = (const float*)d_in[3];
  const float* Wr1 = (const float*)d_in[4];
  const float* Wl2 = (const float*)d_in[5];
  const float* bl2 = (const float*)d_in[6];
  const float* Wr2 = (const float*)d_in[7];
  float* out = (float*)d_out;

  const long long NF = (long long)NNODE * INC;   // 6.4M elements

  // Workspace layout: f32 region then f16 region.
  float* deg = (float*)d_ws;                     // N f32
  float* agg = deg + NNODE;                      // N*128 f32
  _Float16* xh    = (_Float16*)(agg + NF);       // N*128 f16
  _Float16* hh    = xh + NF;                     // N*128 f16
  _Float16* aggh  = hh + NF;                     // N*128 f16
  _Float16* Wl1h  = aggh + NF;                   // 16384 f16
  _Float16* Wr1h  = Wl1h + 128 * 128;
  _Float16* Wl2h  = Wr1h + 128 * 128;            // 8192 f16
  _Float16* Wr2h  = Wl2h + 64 * 128;

  const int T = 256;
  #define GRID1(n) ((unsigned)(((long long)(n) + T - 1) / T))

  // Zero deg + agg (contiguous); one-time f16 conversions of x and weights.
  zero_kernel<<<GRID1(NNODE + NF), T, 0, stream>>>(deg, NNODE + NF);
  cvt_f16_kernel<<<GRID1(NF), T, 0, stream>>>(x, xh, NF);
  cvt_f16_kernel<<<GRID1(128 * 128), T, 0, stream>>>(Wl1, Wl1h, 128 * 128);
  cvt_f16_kernel<<<GRID1(128 * 128), T, 0, stream>>>(Wr1, Wr1h, 128 * 128);
  cvt_f16_kernel<<<GRID1(64 * 128), T, 0, stream>>>(Wl2, Wl2h, 64 * 128);
  cvt_f16_kernel<<<GRID1(64 * 128), T, 0, stream>>>(Wr2, Wr2h, 64 * 128);

  degree_kernel<<<GRID1(NEDGE), T, 0, stream>>>(dstp, deg, NEDGE);
  deginv_kernel<<<GRID1(NNODE), T, 0, stream>>>(deg, NNODE);

  // Layer 1: gather/scatter-mean of xh, convert agg, fused dual-GEMM -> hh
  scatter_h16_kernel<<<GRID1((long long)NEDGE * 32), T, 0, stream>>>(
      xh, src, dstp, agg, NEDGE);
  cvt_agg_kernel<<<GRID1(NF), T, 0, stream>>>(agg, deg, aggh, NF);
  sage_wmma_kernel<<<NNODE / 16, 256, 0, stream>>>(
      aggh, xh, Wl1h, bl1, Wr1h, nullptr, hh, INC, 1);

  // Layer 2: re-zero agg, gather/scatter-mean of hh, fused dual-GEMM -> out
  zero_kernel<<<GRID1(NF), T, 0, stream>>>(agg, NF);
  scatter_h16_kernel<<<GRID1((long long)NEDGE * 32), T, 0, stream>>>(
      hh, src, dstp, agg, NEDGE);
  cvt_agg_kernel<<<GRID1(NF), T, 0, stream>>>(agg, deg, aggh, NF);
  sage_wmma_kernel<<<NNODE / 16, 128, 0, stream>>>(
      aggh, hh, Wl2h, bl2, Wr2h, out, nullptr, 64, 0);
}
